// BloomAttention_32804960206883
// MI455X (gfx1250) — matrix operand order repaired and checked by reading, not compiled, not run on hardware
//
#include <hip/hip_runtime.h>
#include <hip/hip_bf16.h>
#include <cstdint>

// ---------------------------------------------------------------------------
// Bloom attention block on gfx1250: all GEMMs via v_wmma_f32_16x16x32_bf16.
// Round 3: B fragments get distinct registers per k-step so the scheduler can
// stagger s_wait_loadcnt instead of fully draining before each WMMA pair.
// ---------------------------------------------------------------------------

#define HID    2048
#define NH     16
#define HD     128
#define SEQ    2048
#define BATCH  2
#define MTOT   (BATCH * SEQ)     // 4096 rows of activations
#define NQKV   (3 * HID)         // 6144

typedef __attribute__((ext_vector_type(16))) __bf16 v16bf;
typedef __attribute__((ext_vector_type(8)))  __bf16 v8bf;
typedef __attribute__((ext_vector_type(8)))  float  v8f;

// -------------------- WMMA fragment helpers (wave32) -----------------------
// A (16x32 bf16): lane&15 = row M; lanes 0-15 hold K = {0..7, 16..23},
//                 lanes 16-31 hold K = {8..15, 24..31} (ISA 7.12.2).
__device__ __forceinline__ v16bf load_a_frag(const __bf16* __restrict__ base,
                                             int ld, int row0, int k0, int lane) {
  const int r  = row0 + (lane & 15);
  const int kk = k0 + ((lane >> 4) << 3);
  v8bf lo = *(const v8bf*)(base + (size_t)r * ld + kk);
  v8bf hi = *(const v8bf*)(base + (size_t)r * ld + kk + 16);
  v16bf f;
#pragma unroll
  for (int i = 0; i < 8; ++i) { f[i] = lo[i]; f[i + 8] = hi[i]; }
  return f;
}

// B (32x16 bf16): lane&15 = column N; lanes 0-15 hold K = {0..7, 16..23},
//                 lanes 16-31 hold K = {8..15, 24..31} (same order as A).
// "column n of B over K" == row (col0+n) of a row-major [N,K] matrix.
__device__ __forceinline__ v16bf load_b_frag(const __bf16* __restrict__ base,
                                             int ld, int col0, int k0, int lane) {
  const int c  = col0 + (lane & 15);
  const int kk = k0 + ((lane >> 4) << 3);
  v8bf lo = *(const v8bf*)(base + (size_t)c * ld + kk);
  v8bf hi = *(const v8bf*)(base + (size_t)c * ld + kk + 16);
  v16bf f;
#pragma unroll
  for (int i = 0; i < 8; ++i) { f[i] = lo[i]; f[i + 8] = hi[i]; }
  return f;
}

__device__ __forceinline__ v8f wmma_bf16(v16bf a, v16bf b, v8f c) {
  return __builtin_amdgcn_wmma_f32_16x16x32_bf16(false, a, false, b,
                                                 (short)0, c, false, false);
}

// -------------------- pass 1: fp32 -> bf16 conversion ----------------------
__global__ void cvt_f32_bf16_kernel(const float* __restrict__ src,
                                    __bf16* __restrict__ dst, int n) {
  int i = blockIdx.x * blockDim.x + threadIdx.x;
  int stride = gridDim.x * blockDim.x;
  for (; i < n; i += stride) dst[i] = (__bf16)src[i];
}

// -------------------- pass 2: QKV GEMM + scatter ---------------------------
// C[M=4096, N=6144] = X @ Wqkv^T + b.  Block = 8 waves = 64M x 256N tile;
// each wave does 32M x 64N: 2 A frags + 4 B frags -> 8 WMMAs per k-step
// (12 b128 loads / 8 WMMAs, all loads issued before first WMMA).
__global__ __launch_bounds__(256) void qkv_gemm_kernel(
    const __bf16* __restrict__ X, const __bf16* __restrict__ W,
    const float* __restrict__ bias,
    __bf16* __restrict__ Qb, __bf16* __restrict__ Kb, __bf16* __restrict__ Vt) {
  const int lane  = threadIdx.x & 31;
  const int wave  = threadIdx.x >> 5;
  const int mbase = blockIdx.y * 64 + (wave & 1) * 32;
  const int nbase = blockIdx.x * 256 + (wave >> 1) * 64;

  v8f acc[2][4] = {{{}, {}, {}, {}}, {{}, {}, {}, {}}};
  for (int k0 = 0; k0 < HID; k0 += 32) {
    if ((k0 & 255) == 0 && k0 + 256 < HID) {   // prefetch next K-block (B rows)
#pragma unroll
      for (int t = 0; t < 4; ++t)
        __builtin_prefetch(W + (size_t)(nbase + 16 * t + (lane & 15)) * HID +
                               k0 + 256, 0, 0);
    }
    v16bf a0 = load_a_frag(X, HID, mbase,      k0, lane);
    v16bf a1 = load_a_frag(X, HID, mbase + 16, k0, lane);
    v16bf b0 = load_b_frag(W, HID, nbase,      k0, lane);
    v16bf b1 = load_b_frag(W, HID, nbase + 16, k0, lane);
    v16bf b2 = load_b_frag(W, HID, nbase + 32, k0, lane);
    v16bf b3 = load_b_frag(W, HID, nbase + 48, k0, lane);
    acc[0][0] = wmma_bf16(a0, b0, acc[0][0]);
    acc[1][0] = wmma_bf16(a1, b0, acc[1][0]);
    acc[0][1] = wmma_bf16(a0, b1, acc[0][1]);
    acc[1][1] = wmma_bf16(a1, b1, acc[1][1]);
    acc[0][2] = wmma_bf16(a0, b2, acc[0][2]);
    acc[1][2] = wmma_bf16(a1, b2, acc[1][2]);
    acc[0][3] = wmma_bf16(a0, b3, acc[0][3]);
    acc[1][3] = wmma_bf16(a1, b3, acc[1][3]);
  }

  const int c    = lane & 15;
  const int half = lane >> 4;
#pragma unroll
  for (int mt = 0; mt < 2; ++mt) {
#pragma unroll
    for (int t = 0; t < 4; ++t) {
      const int n     = nbase + 16 * t + c;
      const int head  = n / (3 * HD);
      const int rem   = n - head * (3 * HD);
      const int which = rem / HD;          // 0=Q 1=K 2=V (uniform per sub-tile)
      const int d     = rem - which * HD;
      const float bn  = bias[n];
      const int m0    = mbase + mt * 16 + half * 8;  // 8 consecutive rows
      const int batch = m0 >> 11;
      const int s0    = m0 & (SEQ - 1);
      const int bh    = batch * NH + head;
      if (which == 2) {
        v8bf vv;
#pragma unroll
        for (int r = 0; r < 8; ++r) vv[r] = (__bf16)(acc[mt][t][r] + bn);
        *(v8bf*)(Vt + ((size_t)bh * HD + d) * SEQ + s0) = vv;   // 16B store
      } else {
        __bf16* dst = (which == 0) ? Qb : Kb;
#pragma unroll
        for (int r = 0; r < 8; ++r)
          dst[((size_t)bh * SEQ + (s0 + r)) * HD + d] =
              (__bf16)(acc[mt][t][r] + bn);
      }
    }
  }
}

// -------------------- pass 3: flash attention ------------------------------
// One wave = one (batch, head, 16-row q tile); consecutive waves in a block
// share (batch, head) so K/V tiles hit the WGP cache.  Online softmax fp32.
__global__ __launch_bounds__(128) void flash_attn_kernel(
    const __bf16* __restrict__ Qb, const __bf16* __restrict__ Kb,
    const __bf16* __restrict__ Vt, const float* __restrict__ amask,
    __bf16* __restrict__ Ctx) {
  __shared__ alignas(16) __bf16 plds[4][16 * 32];   // P re-layout buffer

  const int lane  = threadIdx.x & 31;
  const int wave  = threadIdx.x >> 5;
  const int w     = blockIdx.x * 4 + wave;
  const int bh    = w >> 7;                 // / (SEQ/16)
  const int qtile = w & 127;
  const int batch = bh >> 4;
  const int head  = bh & 15;
  const int qbase = qtile * 16;
  const float slope = exp2f(-8.0f * (float)(head + 1) / 16.0f);
  const float alpha = 0.08838834764831845f; // 1/sqrt(128); layer factors cancel

  const __bf16* Qp = Qb + (size_t)bh * SEQ * HD;
  const __bf16* Kp = Kb + (size_t)bh * SEQ * HD;
  const __bf16* Vp = Vt + (size_t)bh * HD * SEQ;

  v16bf qf[4];
#pragma unroll
  for (int sl = 0; sl < 4; ++sl) qf[sl] = load_a_frag(Qp, HD, qbase, sl * 32, lane);

  v8f acc[8] = {};                          // 16 x 128 context, fp32
  float mrow[8], lrow[8];
#pragma unroll
  for (int r = 0; r < 8; ++r) { mrow[r] = -1e30f; lrow[r] = 0.f; }

  const int c = lane & 15, half = lane >> 4;
  const int ktiles = (qbase + 15) / 32 + 1;           // causal upper bound

  for (int kt = 0; kt < ktiles; ++kt) {
    const int k0 = kt * 32;
    const bool diag = (k0 + 31) > qbase;              // needs causal masking
    v8f s0 = {}, s1 = {};
#pragma unroll
    for (int sl = 0; sl < 4; ++sl) {                  // QK^T, contraction = hd
      v16bf b0 = load_b_frag(Kp, HD, k0,      sl * 32, lane);
      v16bf b1 = load_b_frag(Kp, HD, k0 + 16, sl * 32, lane);
      s0 = wmma_bf16(qf[sl], b0, s0);
      s1 = wmma_bf16(qf[sl], b1, s1);
    }
    const int kidx0 = k0 + c, kidx1 = k0 + 16 + c;
    const float al0 = slope * (float)kidx0, al1 = slope * (float)kidx1;
    const bool kv0 = amask[batch * SEQ + kidx0] != 0.f;
    const bool kv1 = amask[batch * SEQ + kidx1] != 0.f;

    float scale[8];
#pragma unroll
    for (int r = 0; r < 8; ++r) {
      if (diag) {
        const int q = qbase + half * 8 + r;
        s0[r] = (kv0 && kidx0 <= q) ? (alpha * s0[r] + al0) : -1e30f;
        s1[r] = (kv1 && kidx1 <= q) ? (alpha * s1[r] + al1) : -1e30f;
      } else {
        s0[r] = kv0 ? (alpha * s0[r] + al0) : -1e30f;
        s1[r] = kv1 ? (alpha * s1[r] + al1) : -1e30f;
      }
      // row reductions stay inside each 16-lane half (offsets <= 8)
      float mx = fmaxf(s0[r], s1[r]);
#pragma unroll
      for (int off = 8; off; off >>= 1) mx = fmaxf(mx, __shfl_xor(mx, off, 32));
      const float mnew = fmaxf(mrow[r], mx);
      const float sc = __expf(mrow[r] - mnew);
      s0[r] = __expf(s0[r] - mnew);
      s1[r] = __expf(s1[r] - mnew);
      float sum = s0[r] + s1[r];
#pragma unroll
      for (int off = 8; off; off >>= 1) sum += __shfl_xor(sum, off, 32);
      lrow[r] = lrow[r] * sc + sum;
      mrow[r] = mnew;
      scale[r] = sc;
    }
#pragma unroll
    for (int j = 0; j < 8; ++j)
#pragma unroll
      for (int r = 0; r < 8; ++r) acc[j][r] *= scale[r];

    // C-layout -> A-layout for P via LDS (wave-private; DS ops are in-order
    // per wave, so a dscnt wait suffices — no cross-wave barrier legal here).
    __bf16* pb = plds[wave];
#pragma unroll
    for (int r = 0; r < 8; ++r) {
      const int m = half * 8 + r;
      pb[m * 32 + c]      = (__bf16)s0[r];
      pb[m * 32 + 16 + c] = (__bf16)s1[r];
    }
    asm volatile("s_wait_dscnt 0" ::: "memory");
    v16bf pf;
    {
      const int row = lane & 15;
      const int kk  = (lane >> 4) << 3;
      v8bf lo = *(const v8bf*)(pb + row * 32 + kk);
      v8bf hi = *(const v8bf*)(pb + row * 32 + kk + 16);
#pragma unroll
      for (int i = 0; i < 8; ++i) { pf[i] = lo[i]; pf[i + 8] = hi[i]; }
    }
#pragma unroll
    for (int j = 0; j < 8; ++j) {                     // PV, contraction = k
      v16bf bv = load_b_frag(Vp, SEQ, j * 16, k0, lane);
      acc[j] = wmma_bf16(pf, bv, acc[j]);
    }
  }

#pragma unroll
  for (int r = 0; r < 8; ++r) lrow[r] = 1.0f / lrow[r];
#pragma unroll
  for (int j = 0; j < 8; ++j)
#pragma unroll
    for (int r = 0; r < 8; ++r) {
      const int s = qbase + half * 8 + r;
      Ctx[((size_t)(batch * SEQ + s)) * HID + head * HD + j * 16 + c] =
          (__bf16)(acc[j][r] * lrow[r]);
    }
}

// -------------------- pass 4: dense GEMM + bias + residual -----------------
__global__ __launch_bounds__(256) void dense_gemm_kernel(
    const __bf16* __restrict__ Ctx, const __bf16* __restrict__ W,
    const float* __restrict__ bias, const float* __restrict__ residual,
    float* __restrict__ out) {
  const int lane  = threadIdx.x & 31;
  const int wave  = threadIdx.x >> 5;
  const int mbase = blockIdx.y * 64 + (wave & 1) * 32;
  const int nbase = blockIdx.x * 256 + (wave >> 1) * 64;

  v8f acc[2][4] = {{{}, {}, {}, {}}, {{}, {}, {}, {}}};
  for (int k0 = 0; k0 < HID; k0 += 32) {
    if ((k0 & 255) == 0 && k0 + 256 < HID) {
#pragma unroll
      for (int t = 0; t < 4; ++t)
        __builtin_prefetch(W + (size_t)(nbase + 16 * t + (lane & 15)) * HID +
                               k0 + 256, 0, 0);
    }
    v16bf a0 = load_a_frag(Ctx, HID, mbase,      k0, lane);
    v16bf a1 = load_a_frag(Ctx, HID, mbase + 16, k0, lane);
    v16bf b0 = load_b_frag(W, HID, nbase,      k0, lane);
    v16bf b1 = load_b_frag(W, HID, nbase + 16, k0, lane);
    v16bf b2 = load_b_frag(W, HID, nbase + 32, k0, lane);
    v16bf b3 = load_b_frag(W, HID, nbase + 48, k0, lane);
    acc[0][0] = wmma_bf16(a0, b0, acc[0][0]);
    acc[1][0] = wmma_bf16(a1, b0, acc[1][0]);
    acc[0][1] = wmma_bf16(a0, b1, acc[0][1]);
    acc[1][1] = wmma_bf16(a1, b1, acc[1][1]);
    acc[0][2] = wmma_bf16(a0, b2, acc[0][2]);
    acc[1][2] = wmma_bf16(a1, b2, acc[1][2]);
    acc[0][3] = wmma_bf16(a0, b3, acc[0][3]);
    acc[1][3] = wmma_bf16(a1, b3, acc[1][3]);
  }
  const int c = lane & 15, half = lane >> 4;
#pragma unroll
  for (int mt = 0; mt < 2; ++mt) {
#pragma unroll
    for (int t = 0; t < 4; ++t) {
      const int n = nbase + 16 * t + c;
      const float bn = bias[n];
#pragma unroll
      for (int r = 0; r < 8; ++r) {
        const int m = mbase + mt * 16 + half * 8 + r;
        out[(size_t)m * HID + n] =
            acc[mt][t][r] + bn + residual[(size_t)m * HID + n];
      }
    }
  }
}

// -------------------- launch ------------------------------------------------
extern "C" void kernel_launch(void* const* d_in, const int* in_sizes, int n_in,
                              void* d_out, int out_size, void* d_ws, size_t ws_size,
                              hipStream_t stream) {
  const float* hs   = (const float*)d_in[0];   // [2,2048,2048]
  const float* res  = (const float*)d_in[1];   // [2,2048,2048]
  const float* am   = (const float*)d_in[2];   // [2,2048]
  const float* Wqkv = (const float*)d_in[3];   // [6144,2048]
  const float* bqkv = (const float*)d_in[4];   // [6144]
  const float* Wd   = (const float*)d_in[5];   // [2048,2048]
  const float* bd   = (const float*)d_in[6];   // [2048]
  float* out        = (float*)d_out;

  char* ws = (char*)d_ws;
  size_t off = 0;
  auto take = [&](size_t bytes) { char* p = ws + off; off += (bytes + 255) & ~(size_t)255; return p; };
  __bf16* Xbf  = (__bf16*)take((size_t)MTOT * HID * 2);
  __bf16* Wqb  = (__bf16*)take((size_t)NQKV * HID * 2);
  __bf16* Wdb  = (__bf16*)take((size_t)HID * HID * 2);
  __bf16* Qb   = (__bf16*)take((size_t)BATCH * NH * SEQ * HD * 2);
  __bf16* Kb   = (__bf16*)take((size_t)BATCH * NH * SEQ * HD * 2);
  __bf16* Vt   = (__bf16*)take((size_t)BATCH * NH * HD * SEQ * 2);
  __bf16* Ctx  = (__bf16*)take((size_t)MTOT * HID * 2);
  (void)ws_size; (void)in_sizes; (void)n_in; (void)out_size;

  cvt_f32_bf16_kernel<<<2048, 256, 0, stream>>>(hs,   Xbf, MTOT * HID);
  cvt_f32_bf16_kernel<<<2048, 256, 0, stream>>>(Wqkv, Wqb, NQKV * HID);
  cvt_f32_bf16_kernel<<<2048, 256, 0, stream>>>(Wd,   Wdb, HID * HID);

  dim3 gq(NQKV / 256, MTOT / 64);
  qkv_gemm_kernel<<<gq, 256, 0, stream>>>(Xbf, Wqb, bqkv, Qb, Kb, Vt);

  flash_attn_kernel<<<(BATCH * NH * (SEQ / 16)) / 4, 128, 0, stream>>>(
      Qb, Kb, Vt, am, Ctx);

  dim3 gd(HID / 256, MTOT / 64);
  dense_gemm_kernel<<<gd, 256, 0, stream>>>(Ctx, Wdb, bd, res, out);
}
